// LearnableNormalizedGaussianAndStudentAffinity_82102594830652
// MI455X (gfx1250) — matrix operand order repaired and checked by reading, not compiled, not run on hardware
//
#include <hip/hip_runtime.h>
#include <hip/hip_bf16.h>

// ---------------------------------------------------------------------------
// Problem constants (reference: X [8192,1024] fp32, output [8192,8192] fp32)
// ---------------------------------------------------------------------------
#define N_PTS 8192
#define KDIM  1024

typedef __attribute__((ext_vector_type(16))) __bf16 v16bf;
typedef __attribute__((ext_vector_type(8)))  __bf16 v8bf;
typedef __attribute__((ext_vector_type(8)))  float  v8f;

// ---------------- bf16 helpers (round-to-nearest-even split) ---------------
static __device__ __forceinline__ __bf16 bf16_rn(float f) {
    union { float f; unsigned u; } in; in.f = f;
    unsigned u = in.u + 0x7fffu + ((in.u >> 16) & 1u);
    union { unsigned short s; __bf16 b; } out; out.s = (unsigned short)(u >> 16);
    return out.b;
}
static __device__ __forceinline__ float bf16_f32(__bf16 b) {
    union { __bf16 b; unsigned short s; } in; in.b = b;
    union { unsigned u; float f; } out; out.u = ((unsigned)in.s) << 16;
    return out.f;
}

// ---------------------------------------------------------------------------
// Pass 0: row norms + hi/lo bf16 split of X; reset global accumulator.
// One 256-thread block per row (1024 cols -> 4 per thread).
// ---------------------------------------------------------------------------
__global__ __launch_bounds__(256)
void prep_kernel(const float* __restrict__ X, float* __restrict__ sq,
                 __bf16* __restrict__ Xhi, __bf16* __restrict__ Xlo,
                 double* __restrict__ accum) {
    const int row = blockIdx.x;
    const int t   = threadIdx.x;
    if (row == 0 && t == 0) *accum = 0.0;

    const float* xr = X + (size_t)row * KDIM;
    float local = 0.0f;
    const int k0 = t * 4;
    float4 v = *(const float4*)(xr + k0);
    float vv[4] = {v.x, v.y, v.z, v.w};
    #pragma unroll
    for (int e = 0; e < 4; ++e) {
        float x = vv[e];
        local = fmaf(x, x, local);
        __bf16 h = bf16_rn(x);
        Xhi[(size_t)row * KDIM + k0 + e] = h;
        Xlo[(size_t)row * KDIM + k0 + e] = bf16_rn(x - bf16_f32(h));
    }
    __shared__ float red[256];
    red[t] = local;
    __syncthreads();
    #pragma unroll
    for (int s = 128; s > 0; s >>= 1) {
        if (t < s) red[t] += red[t + s];
        __syncthreads();
    }
    if (t == 0) sq[row] = red[0];
}

// ---------------------------------------------------------------------------
// WMMA fragment loaders, per CDNA5 ISA 16-bit layouts (CDNA5 ISA).
// A 16x32 (MxK): lane l<16 -> row l,    K = {0..7} U {16..23}
//                lane l>=16 -> row l-16, K = {8..15} U {24..31}
// B 32x16 (KxN): lane l<16 -> col l,    K = {0..7} U {16..23}
//                lane l>=16 -> col l-16, K = {8..15} U {24..31}
// Since B = X^T, "column n" of B is row n of X -> same row-major source.
// ---------------------------------------------------------------------------
static __device__ __forceinline__ v16bf load_a_frag(const __bf16* __restrict__ M,
                                                    int row_base, int k_base, int lane) {
    const __bf16* p = M + (size_t)(row_base + (lane & 15)) * KDIM
                        + k_base + ((lane >> 4) << 3);
    v8bf lo = *(const v8bf*)p;          // K offsets +0..7
    v8bf hi = *(const v8bf*)(p + 16);   // K offsets +16..23
    return __builtin_shufflevector(lo, hi, 0,1,2,3,4,5,6,7,8,9,10,11,12,13,14,15);
}
static __device__ __forceinline__ v16bf load_b_frag(const __bf16* __restrict__ M,
                                                    int col_base, int k_base, int lane) {
    const __bf16* p = M + (size_t)(col_base + (lane & 15)) * KDIM
                        + k_base + ((lane >> 4) << 3);
    v8bf lo = *(const v8bf*)p;          // K offsets +0..7
    v8bf hi = *(const v8bf*)(p + 16);   // K offsets +16..23
    return __builtin_shufflevector(lo, hi, 0,1,2,3,4,5,6,7,8,9,10,11,12,13,14,15);
}

static __device__ __forceinline__ v8f wmma_bf16(v16bf a, v16bf b, v8f c) {
    return __builtin_amdgcn_wmma_f32_16x16x32_bf16(
        /*neg_a=*/false, a, /*neg_b=*/false, b,
        /*c_mod=*/(short)0, c, /*reuse_a=*/false, /*reuse_b=*/false);
}

// ---------------------------------------------------------------------------
// Pass 1: split-bf16 WMMA GEMM (dot ~= hi*hi + hi*lo + lo*hi), fused Student-t
// log-density epilogue, per-block partial sum of exp(log_P) -> f64 atomic.
// Block = 256 threads = 8 waves; wave tile 32x32 (2x2 WMMA); block tile 64x128.
// ---------------------------------------------------------------------------
__global__ __launch_bounds__(256)
void gemm_logp_kernel(const __bf16* __restrict__ Xhi, const __bf16* __restrict__ Xlo,
                      const float* __restrict__ sq, const float* __restrict__ alpha_p,
                      float* __restrict__ out, double* __restrict__ accum) {
    const int lane = threadIdx.x & 31;
    const int wave = threadIdx.x >> 5;
    const int tileRow = blockIdx.y * 64  + (wave >> 2) * 32;  // 2 wave-rows
    const int tileCol = blockIdx.x * 128 + (wave & 3) * 32;   // 4 wave-cols

    v8f acc[2][2];
    #pragma unroll
    for (int i = 0; i < 2; ++i)
        #pragma unroll
        for (int j = 0; j < 2; ++j)
            acc[i][j] = (v8f)(0.0f);

    for (int k = 0; k < KDIM; k += 32) {
        if (k + 32 < KDIM) {
            // High-locality prefetch of next A/B k-blocks -> WGP-scope
            // global_prefetch_b8 (operands are L2-resident; pull into near cache).
            __builtin_prefetch(Xhi + (size_t)(tileRow + (lane & 15)) * KDIM + k + 32, 0, 3);
            __builtin_prefetch(Xhi + (size_t)(tileCol + (lane & 15)) * KDIM + k + 32, 0, 3);
        }
        v16bf ah0 = load_a_frag(Xhi, tileRow,      k, lane);
        v16bf ah1 = load_a_frag(Xhi, tileRow + 16, k, lane);
        v16bf al0 = load_a_frag(Xlo, tileRow,      k, lane);
        v16bf al1 = load_a_frag(Xlo, tileRow + 16, k, lane);
        v16bf bh0 = load_b_frag(Xhi, tileCol,      k, lane);
        v16bf bh1 = load_b_frag(Xhi, tileCol + 16, k, lane);
        v16bf bl0 = load_b_frag(Xlo, tileCol,      k, lane);
        v16bf bl1 = load_b_frag(Xlo, tileCol + 16, k, lane);

        // hi*hi + hi*lo + lo*hi  (lo*lo term below fp32 ulp; dropped)
        acc[0][0] = wmma_bf16(ah0, bh0, acc[0][0]);
        acc[0][1] = wmma_bf16(ah0, bh1, acc[0][1]);
        acc[1][0] = wmma_bf16(ah1, bh0, acc[1][0]);
        acc[1][1] = wmma_bf16(ah1, bh1, acc[1][1]);

        acc[0][0] = wmma_bf16(ah0, bl0, acc[0][0]);
        acc[0][1] = wmma_bf16(ah0, bl1, acc[0][1]);
        acc[1][0] = wmma_bf16(ah1, bl0, acc[1][0]);
        acc[1][1] = wmma_bf16(ah1, bl1, acc[1][1]);

        acc[0][0] = wmma_bf16(al0, bh0, acc[0][0]);
        acc[0][1] = wmma_bf16(al0, bh1, acc[0][1]);
        acc[1][0] = wmma_bf16(al1, bh0, acc[1][0]);
        acc[1][1] = wmma_bf16(al1, bh1, acc[1][1]);
    }

    // Student-t epilogue.  nu = softplus(alpha); log_P = -((nu+1)/2)*log1p(C/nu)
    const float a     = alpha_p[0];
    const float nu    = (a > 20.0f) ? a : log1pf(__expf(a));
    const float coef  = -0.5f * (nu + 1.0f);
    const float invnu = 1.0f / nu;

    float psum = 0.0f;
    // C/D layout: VGPR r, lane l -> M = r + (l>>4)*8, N = l&15
    #pragma unroll
    for (int i = 0; i < 2; ++i) {
        #pragma unroll
        for (int j = 0; j < 2; ++j) {
            const int colN = tileCol + j * 16 + (lane & 15);
            const float sqc = sq[colN];
            #pragma unroll
            for (int r = 0; r < 8; ++r) {
                const int rowM = tileRow + i * 16 + ((lane >> 4) << 3) + r;
                float cval = sq[rowM] + sqc - 2.0f * acc[i][j][r];
                cval = fmaxf(cval, 0.0f);
                const float lp = coef * log1pf(cval * invnu);
                out[(size_t)rowM * N_PTS + colN] = lp;
                psum += __expf(lp);   // max-shift is 0 (diagonal has log_P = 0)
            }
        }
    }

    __shared__ float red[256];
    red[threadIdx.x] = psum;
    __syncthreads();
    #pragma unroll
    for (int s = 128; s > 0; s >>= 1) {
        if (threadIdx.x < s) red[threadIdx.x] += red[threadIdx.x + s];
        __syncthreads();
    }
    if (threadIdx.x == 0) atomicAdd(accum, (double)red[0]);
}

// ---------------------------------------------------------------------------
// Pass 2: subtract global logsumexp = log(sum exp(log_P)) from every element.
// ---------------------------------------------------------------------------
__global__ __launch_bounds__(256)
void normalize_kernel(float* __restrict__ out, const double* __restrict__ accum,
                      size_t n4) {
    const float lse = (float)log(*accum);
    size_t i = (size_t)blockIdx.x * blockDim.x + threadIdx.x;
    const size_t stride = (size_t)gridDim.x * blockDim.x;
    for (; i < n4; i += stride) {
        float4 v = ((const float4*)out)[i];
        v.x -= lse; v.y -= lse; v.z -= lse; v.w -= lse;
        ((float4*)out)[i] = v;
    }
}

// ---------------------------------------------------------------------------
// Launch.  Workspace layout:
//   [0, 8)                       : f64 accumulator
//   [256, 256 + 32KB)            : sq (8192 f32 row norms)
//   then Xhi (16 MB bf16), Xlo (16 MB bf16)
// ---------------------------------------------------------------------------
extern "C" void kernel_launch(void* const* d_in, const int* in_sizes, int n_in,
                              void* d_out, int out_size, void* d_ws, size_t ws_size,
                              hipStream_t stream) {
    (void)in_sizes; (void)n_in; (void)out_size; (void)ws_size;
    const float* X     = (const float*)d_in[0];
    const float* alpha = (const float*)d_in[1];
    float*       out   = (float*)d_out;

    char*   ws    = (char*)d_ws;
    double* accum = (double*)ws;
    float*  sq    = (float*)(ws + 256);
    __bf16* Xhi   = (__bf16*)(ws + 256 + N_PTS * sizeof(float));
    __bf16* Xlo   = Xhi + (size_t)N_PTS * KDIM;

    prep_kernel<<<N_PTS, 256, 0, stream>>>(X, sq, Xhi, Xlo, accum);

    dim3 grid(N_PTS / 128, N_PTS / 64);
    gemm_logp_kernel<<<grid, 256, 0, stream>>>(Xhi, Xlo, sq, alpha, out, accum);

    const size_t n4 = (size_t)N_PTS * N_PTS / 4;
    normalize_kernel<<<8192, 256, 0, stream>>>(out, accum, n4);
}
